// LstmAttentionModel_48490180772327
// MI455X (gfx1250) — matrix operand order repaired and checked by reading, not compiled, not run on hardware
//
#include <hip/hip_runtime.h>
#include <math.h>

// ---------------- CDNA5 WMMA types ----------------
typedef __attribute__((ext_vector_type(16))) __bf16 bf16x16;
typedef __attribute__((ext_vector_type(8)))  float  f32x8;
typedef __attribute__((ext_vector_type(4)))  int    i32x4;

union Frag { bf16x16 v; uint4 q[2]; };

#define AS1 __attribute__((address_space(1)))
#define AS3 __attribute__((address_space(3)))

#if defined(__has_builtin)
#if __has_builtin(__builtin_amdgcn_global_load_async_to_lds_b128)
#define USE_ASYNC_LDS 1
#endif
#endif
#ifndef USE_ASYNC_LDS
#define USE_ASYNC_LDS 0
#endif

__device__ __forceinline__ void copy16_g2l(const unsigned short* g, unsigned short* l) {
#if USE_ASYNC_LDS
  unsigned short* gnc = const_cast<unsigned short*>(g);
  __builtin_amdgcn_global_load_async_to_lds_b128((AS1 i32x4*)gnc, (AS3 i32x4*)l, 0, 0);
#else
  *(uint4*)l = *(const uint4*)g;
#endif
}
__device__ __forceinline__ void wait_async_cnt() {
#if USE_ASYNC_LDS
#if __has_builtin(__builtin_amdgcn_s_wait_asynccnt)
  __builtin_amdgcn_s_wait_asynccnt(0);
#endif
#endif
}

__device__ __forceinline__ unsigned short f2bf(float x) {
  unsigned int u = __float_as_uint(x);
  u += 0x7FFFu + ((u >> 16) & 1u);   // round-to-nearest-even bf16
  return (unsigned short)(u >> 16);
}
__device__ __forceinline__ float sigf(float x) { return 1.f / (1.f + __expf(-x)); }
__device__ __forceinline__ f32x8 v8zero() {
  f32x8 z = {0.f,0.f,0.f,0.f,0.f,0.f,0.f,0.f}; return z;
}
__device__ __forceinline__ f32x8 wmma_bf16(const Frag& a, const Frag& b, f32x8 c) {
  return __builtin_amdgcn_wmma_f32_16x16x32_bf16(false, a.v, false, b.v,
                                                 (short)0, c, false, false);
}

// Model dims (fixed by the reference)
// B=16, TE=TD=128, E=512, H=512, D=1024, V=32000

// ---------------- elementwise helpers ----------------
__global__ void k_cvt_bf16(const float* __restrict__ s, unsigned short* __restrict__ d, long n) {
  long i = (long)blockIdx.x * blockDim.x + threadIdx.x;
  long st = (long)gridDim.x * blockDim.x;
  for (; i < n; i += st) d[i] = f2bf(s[i]);
}

__global__ void k_addv(const float* __restrict__ a, const float* __restrict__ b,
                       float* __restrict__ o, int n) {
  int i = blockIdx.x * 256 + threadIdx.x;
  if (i < n) o[i] = a[i] + b[i];
}

// gather embeddings -> bf16 rows (E = 512 fixed)
__global__ void k_embed(const int* __restrict__ seq, const float* __restrict__ emb,
                        unsigned short* __restrict__ out, int rows) {
  long n = (long)rows * 512;
  long i = (long)blockIdx.x * blockDim.x + threadIdx.x;
  long st = (long)gridDim.x * blockDim.x;
  for (; i < n; i += st) {
    int r = (int)(i >> 9);
    int e = (int)(i & 511);
    out[i] = f2bf(emb[(long)seq[r] * 512 + e]);
  }
}

// vt[b][d][t] = v[(b*128+t)*1024 + d]  (bf16), for ctx GEMM Bt layout
__global__ void k_vt(const float* __restrict__ v, unsigned short* __restrict__ vt) {
  long n = (long)16 * 1024 * 128;
  long i = (long)blockIdx.x * blockDim.x + threadIdx.x;
  long st = (long)gridDim.x * blockDim.x;
  for (; i < n; i += st) {
    int t = (int)(i & 127);
    long rem = i >> 7;
    int d = (int)(rem & 1023);
    int b = (int)(rem >> 10);
    vt[i] = f2bf(v[((long)(b * 128 + t)) * 1024 + d]);
  }
}

// ---------------- generic WMMA GEMM: C = A(MxK) * Bt(NxK)^T + bias ----------------
// Block tile 128x128, 8 wave32s in a 4(M)x2(N) grid, each wave 32x64 (2x4 WMMA frags).
#define GBM 128
#define GBN 128
#define GBK 32

__global__ __launch_bounds__(256)
void k_gemm(const unsigned short* __restrict__ A, const unsigned short* __restrict__ Bt,
            const float* __restrict__ bias, float* __restrict__ C,
            int M, int N, int K, long sA, long sB, long sC) {
  (void)M;
  __shared__ unsigned short Asm[GBM * GBK];   // 8 KB
  __shared__ unsigned short Bsm[GBN * GBK];   // 8 KB
  const int tid = threadIdx.x;
  A  += (long)blockIdx.z * sA;
  Bt += (long)blockIdx.z * sB;
  C  += (long)blockIdx.z * sC;
  const int n0 = blockIdx.x * GBN;
  const int m0 = blockIdx.y * GBM;
  const int w   = tid >> 5, lane = tid & 31;
  const int wm  = w >> 1,   wn   = w & 1;      // wave computes rows wm*32..+32, cols wn*64..+64
  const int grp = lane >> 4, ml  = lane & 15;

  f32x8 acc[2][4];
  #pragma unroll
  for (int mi = 0; mi < 2; ++mi)
    #pragma unroll
    for (int j = 0; j < 4; ++j) acc[mi][j] = v8zero();

  // loaders: 128 rows x 32 halfs per matrix; thread -> one row-half (16 halfs = 2x b128)
  const int lrow = tid >> 1, lkc = (tid & 1) * 16;
  const long aoff = (long)(m0 + lrow) * K + lkc;
  const long boff = (long)(n0 + lrow) * K + lkc;
  unsigned short* adst = &Asm[lrow * GBK + lkc];
  unsigned short* bdst = &Bsm[lrow * GBK + lkc];

  for (int k0 = 0; k0 < K; k0 += GBK) {
    copy16_g2l(&A[aoff + k0],      adst);
    copy16_g2l(&A[aoff + k0 + 8],  adst + 8);
    copy16_g2l(&Bt[boff + k0],     bdst);
    copy16_g2l(&Bt[boff + k0 + 8], bdst + 8);
    wait_async_cnt();
    __syncthreads();
    if (k0 + GBK < K) {                      // gfx1250 global_prefetch of next tiles
      __builtin_prefetch(&A[aoff + k0 + GBK], 0, 0);
      __builtin_prefetch(&Bt[boff + k0 + GBK], 0, 0);
    }
    Frag af0, af1;
    const uint4* ap0 = (const uint4*)&Asm[(wm * 32 + ml) * GBK + grp * 8];
    const uint4* ap1 = (const uint4*)&Asm[(wm * 32 + 16 + ml) * GBK + grp * 8];
    af0.q[0] = ap0[0]; af0.q[1] = ap0[2];    // K packing per ISA 16-bit A layout
    af1.q[0] = ap1[0]; af1.q[1] = ap1[2];
    #pragma unroll
    for (int j = 0; j < 4; ++j) {
      const uint4* bp = (const uint4*)&Bsm[(wn * 64 + j * 16 + ml) * GBK + grp * 8];
      Frag bfr; bfr.q[0] = bp[0]; bfr.q[1] = bp[2];
      acc[0][j] = wmma_bf16(af0, bfr, acc[0][j]);
      acc[1][j] = wmma_bf16(af1, bfr, acc[1][j]);
    }
    __syncthreads();
  }
  const int colb = n0 + wn * 64 + ml;
  const int rowb = m0 + wm * 32 + grp * 8;
  float bv0 = 0.f, bv1 = 0.f, bv2 = 0.f, bv3 = 0.f;
  if (bias) { bv0 = bias[colb]; bv1 = bias[colb + 16]; bv2 = bias[colb + 32]; bv3 = bias[colb + 48]; }
  #pragma unroll
  for (int mi = 0; mi < 2; ++mi) {
    #pragma unroll
    for (int r = 0; r < 8; ++r) {
      long base = (long)(rowb + mi * 16 + r) * N;
      C[base + colb     ] = acc[mi][0][r] + bv0;
      C[base + colb + 16] = acc[mi][1][r] + bv1;
      C[base + colb + 32] = acc[mi][2][r] + bv2;
      C[base + colb + 48] = acc[mi][3][r] + bv3;
    }
  }
}

// ---------------- encoder LSTM (both directions), persistent block per dir ----------------
__global__ __launch_bounds__(256)
void k_enc_lstm(const float* __restrict__ xg_f, const float* __restrict__ xg_b,
                const unsigned short* __restrict__ Whh_f, const unsigned short* __restrict__ Whh_b,
                float* __restrict__ gbuf_all, float* __restrict__ enc_out,
                float* __restrict__ h0o, float* __restrict__ c0o) {
  constexpr int H = 512, GATES = 2048, TE = 128, NB = 16;
  const int dir = blockIdx.x;
  const float* xg = dir ? xg_b : xg_f;
  const unsigned short* Whh = dir ? Whh_b : Whh_f;
  float* gbuf = gbuf_all + (long)dir * NB * GATES;

  __shared__ unsigned short hsm[NB * H];   // 16 KB bf16 h_t
  const int tid = threadIdx.x, w = tid >> 5, lane = tid & 31;
  const int grp = lane >> 4, ml = lane & 15;

  for (int i = tid; i < NB * H; i += 256) hsm[i] = 0;
  float c[32];
  #pragma unroll
  for (int i = 0; i < 32; ++i) c[i] = 0.f;
  __syncthreads();

  for (int t = 0; t < TE; ++t) {
    const int tt = dir ? (TE - 1 - t) : t;
    // phase 1: gates = h @ Whh^T  (M=16 batch rows)
    for (int nt = w; nt < GATES / 16; nt += 8) {
      f32x8 acc = v8zero();
      const unsigned short* bb = Whh + (long)(nt * 16 + ml) * H + grp * 8;
      for (int kk = 0; kk < H; kk += 32) {
        Frag af, bfr;
        const uint4* ap = (const uint4*)&hsm[ml * H + kk + grp * 8];
        af.q[0] = ap[0]; af.q[1] = ap[2];
        const uint4* bp = (const uint4*)&bb[kk];
        bfr.q[0] = bp[0]; bfr.q[1] = bp[2];
        acc = wmma_bf16(af, bfr, acc);
      }
      const int col = nt * 16 + ml;
      #pragma unroll
      for (int r = 0; r < 8; ++r) {
        int b = grp * 8 + r;
        gbuf[b * GATES + col] = acc[r] + xg[((long)(b * TE + tt)) * GATES + col];
      }
    }
    __threadfence_block();
    __syncthreads();
    // phase 2: elementwise gates (torch order i,f,g,o)
    #pragma unroll
    for (int i = 0; i < 32; ++i) {
      int idx = tid + i * 256;          // 0..8191
      int b = idx >> 9, j = idx & 511;
      float gi = gbuf[b * GATES + j];
      float gf = gbuf[b * GATES + H + j];
      float gg = gbuf[b * GATES + 2 * H + j];
      float go = gbuf[b * GATES + 3 * H + j];
      float cn = sigf(gf) * c[i] + sigf(gi) * tanhf(gg);
      c[i] = cn;
      float h = sigf(go) * tanhf(cn);
      hsm[b * H + j] = f2bf(h);
      enc_out[((long)(b * TE + tt)) * 1024 + dir * 512 + j] = h;
      if (t == TE - 1) {
        h0o[b * 1024 + dir * 512 + j] = h;
        c0o[b * 1024 + dir * 512 + j] = cn;
      }
    }
    __threadfence_block();
    __syncthreads();
  }
}

// ---------------- decoder LSTM (hidden = 1024), single persistent block ----------------
__global__ __launch_bounds__(1024)
void k_dec_lstm(const float* __restrict__ xg, const unsigned short* __restrict__ Whh,
                const float* __restrict__ h0, const float* __restrict__ c0,
                float* __restrict__ gbuf, float* __restrict__ dec_out) {
  constexpr int Hd = 1024, GATES = 4096, TD = 128, NB = 16;
  __shared__ unsigned short hsm[NB * Hd];   // 32 KB
  const int tid = threadIdx.x, w = tid >> 5, lane = tid & 31;
  const int grp = lane >> 4, ml = lane & 15;
  float c[16];
  #pragma unroll
  for (int i = 0; i < 16; ++i) {
    int idx = tid + i * 1024;           // 0..16383 == b*1024 + j
    hsm[idx] = f2bf(h0[idx]);
    c[i] = c0[idx];
  }
  __syncthreads();
  for (int t = 0; t < TD; ++t) {
    for (int nt = w; nt < GATES / 16; nt += 32) {
      f32x8 acc = v8zero();
      const unsigned short* bb = Whh + (long)(nt * 16 + ml) * Hd + grp * 8;
      for (int kk = 0; kk < Hd; kk += 32) {
        Frag af, bfr;
        const uint4* ap = (const uint4*)&hsm[ml * Hd + kk + grp * 8];
        af.q[0] = ap[0]; af.q[1] = ap[2];
        const uint4* bp = (const uint4*)&bb[kk];
        bfr.q[0] = bp[0]; bfr.q[1] = bp[2];
        acc = wmma_bf16(af, bfr, acc);
      }
      const int col = nt * 16 + ml;
      #pragma unroll
      for (int r = 0; r < 8; ++r) {
        int b = grp * 8 + r;
        gbuf[b * GATES + col] = acc[r] + xg[((long)(b * TD + t)) * GATES + col];
      }
    }
    __threadfence_block();
    __syncthreads();
    #pragma unroll
    for (int i = 0; i < 16; ++i) {
      int idx = tid + i * 1024;
      int b = idx >> 10, j = idx & 1023;
      float gi = gbuf[b * GATES + j];
      float gf = gbuf[b * GATES + Hd + j];
      float gg = gbuf[b * GATES + 2 * Hd + j];
      float go = gbuf[b * GATES + 3 * Hd + j];
      float cn = sigf(gf) * c[i] + sigf(gi) * tanhf(gg);
      c[i] = cn;
      float h = sigf(go) * tanhf(cn);
      hsm[idx] = f2bf(h);
      dec_out[((long)(b * TD + t)) * Hd + j] = h;
    }
    __threadfence_block();
    __syncthreads();
  }
}

// ---------------- masked softmax over TE=128, scale 1/sqrt(1024) ----------------
__global__ void k_softmax(const float* __restrict__ sim, const int* __restrict__ eseq,
                          unsigned short* __restrict__ attn) {
  constexpr int TE = 128;
  const int row = blockIdx.x;       // b*TD + q
  const int b = row >> 7;
  const int k = threadIdx.x;
  __shared__ float red[128];
  float v = sim[(long)row * TE + k] * (1.f / 32.f);
  if (eseq[b * TE + k] == 0) v = -__builtin_inff();
  red[k] = v; __syncthreads();
  for (int s = 64; s > 0; s >>= 1) { if (k < s) red[k] = fmaxf(red[k], red[k + s]); __syncthreads(); }
  float m = red[0]; __syncthreads();
  float e = __expf(v - m);
  red[k] = e; __syncthreads();
  for (int s = 64; s > 0; s >>= 1) { if (k < s) red[k] += red[k + s]; __syncthreads(); }
  attn[(long)row * TE + k] = f2bf(e / red[0]);
}

// ---------------- tanh(residual) + layernorm -> bf16 ----------------
__global__ __launch_bounds__(256)
void k_ln(const float* __restrict__ dec_out, const float* __restrict__ ctx,
          const float* __restrict__ gamma, const float* __restrict__ beta,
          unsigned short* __restrict__ out_bf) {
  constexpr int D = 1024;
  const int row = blockIdx.x;
  const int tid = threadIdx.x;
  __shared__ float rs[256], rq[256];
  float vals[4]; float s = 0.f, q = 0.f;
  #pragma unroll
  for (int i = 0; i < 4; ++i) {
    int j = tid + i * 256;
    float x = tanhf(dec_out[(long)row * D + j] + ctx[(long)row * D + j]);
    vals[i] = x; s += x; q += x * x;
  }
  rs[tid] = s; rq[tid] = q; __syncthreads();
  for (int st = 128; st > 0; st >>= 1) {
    if (tid < st) { rs[tid] += rs[tid + st]; rq[tid] += rq[tid + st]; }
    __syncthreads();
  }
  float mu = rs[0] / D;
  float var = rq[0] / D - mu * mu;
  float rstd = rsqrtf(var + 1e-5f);
  #pragma unroll
  for (int i = 0; i < 4; ++i) {
    int j = tid + i * 256;
    out_bf[(long)row * D + j] = f2bf((vals[i] - mu) * rstd * gamma[j] + beta[j]);
  }
}

// ---------------- host orchestration ----------------
extern "C" void kernel_launch(void* const* d_in, const int* in_sizes, int n_in,
                              void* d_out, int out_size, void* d_ws, size_t ws_size,
                              hipStream_t stream) {
  (void)in_sizes; (void)n_in; (void)out_size; (void)ws_size;
  const int*   enc_seq = (const int*)  d_in[0];
  const int*   dec_seq = (const int*)  d_in[1];
  const float* enc_emb = (const float*)d_in[2];
  const float* dec_emb = (const float*)d_in[3];
  const float* Wih_f = (const float*)d_in[4];
  const float* Whh_f = (const float*)d_in[5];
  const float* bih_f = (const float*)d_in[6];
  const float* bhh_f = (const float*)d_in[7];
  const float* Wih_b = (const float*)d_in[8];
  const float* Whh_b = (const float*)d_in[9];
  const float* bih_b = (const float*)d_in[10];
  const float* bhh_b = (const float*)d_in[11];
  const float* Wih_d = (const float*)d_in[12];
  const float* Whh_d = (const float*)d_in[13];
  const float* bih_d = (const float*)d_in[14];
  const float* bhh_d = (const float*)d_in[15];
  const float* Wq  = (const float*)d_in[16];
  const float* bq  = (const float*)d_in[17];
  const float* Wk  = (const float*)d_in[18];
  const float* bk  = (const float*)d_in[19];
  const float* Wv  = (const float*)d_in[20];
  const float* bv  = (const float*)d_in[21];
  const float* Wfc = (const float*)d_in[22];
  const float* bfc = (const float*)d_in[23];
  const float* gamma = (const float*)d_in[24];
  const float* beta  = (const float*)d_in[25];
  float* out = (float*)d_out;

  // ---- workspace layout ----
  char* ws = (char*)d_ws;
  size_t off = 0;
  auto alloc = [&](size_t bytes) -> char* {
    off = (off + 255) & ~(size_t)255;
    char* p = ws + off; off += bytes; return p;
  };
  unsigned short* xe_bf   = (unsigned short*)alloc(2048UL * 512 * 2);
  unsigned short* xd_bf   = (unsigned short*)alloc(2048UL * 512 * 2);
  unsigned short* Wihf_bf = (unsigned short*)alloc(2048UL * 512 * 2);
  unsigned short* Whhf_bf = (unsigned short*)alloc(2048UL * 512 * 2);
  unsigned short* Wihb_bf = (unsigned short*)alloc(2048UL * 512 * 2);
  unsigned short* Whhb_bf = (unsigned short*)alloc(2048UL * 512 * 2);
  unsigned short* Wihd_bf = (unsigned short*)alloc(4096UL * 512 * 2);
  unsigned short* Whhd_bf = (unsigned short*)alloc(4096UL * 1024 * 2);
  unsigned short* Wq_bf   = (unsigned short*)alloc(1024UL * 1024 * 2);
  unsigned short* Wk_bf   = (unsigned short*)alloc(1024UL * 1024 * 2);
  unsigned short* Wv_bf   = (unsigned short*)alloc(1024UL * 1024 * 2);
  unsigned short* Wfc_bf  = (unsigned short*)alloc(32000UL * 1024 * 2);
  float* bias_f = (float*)alloc(2048UL * 4);
  float* bias_b = (float*)alloc(2048UL * 4);
  float* bias_d = (float*)alloc(4096UL * 4);
  float* xg_f   = (float*)alloc(2048UL * 2048 * 4);
  float* xg_b   = (float*)alloc(2048UL * 2048 * 4);
  float* xg_d   = (float*)alloc(2048UL * 4096 * 4);
  float* enc_out    = (float*)alloc(2048UL * 1024 * 4);
  unsigned short* enc_out_bf = (unsigned short*)alloc(2048UL * 1024 * 2);
  float* h0buf  = (float*)alloc(16UL * 1024 * 4);
  float* c0buf  = (float*)alloc(16UL * 1024 * 4);
  float* gbuf_enc = (float*)alloc(2UL * 16 * 2048 * 4);
  float* gbuf_dec = (float*)alloc(16UL * 4096 * 4);
  float* dec_out    = (float*)alloc(2048UL * 1024 * 4);
  unsigned short* dec_out_bf = (unsigned short*)alloc(2048UL * 1024 * 2);
  // region reuse (dependencies are stream-ordered):
  float* q_f = xg_f;                                               // after encoder done
  float* k_f = (float*)((char*)xg_f + 2048UL * 1024 * 4);
  float* v_f = xg_b;
  unsigned short* q_bf = (unsigned short*)((char*)xg_b + 2048UL * 1024 * 4);
  unsigned short* k_bf = q_bf + 2048UL * 1024;
  char* p3 = (char*)xg_d;                                          // after decoder done
  unsigned short* vt_bf = (unsigned short*)p3;
  float* sim = (float*)(p3 + 2048UL * 1024 * 2);
  unsigned short* attn_bf = (unsigned short*)((char*)sim + 16UL * 128 * 128 * 4);
  float* ctx = (float*)((char*)attn_bf + 16UL * 128 * 128 * 2 + 256);
  unsigned short* out_bf = (unsigned short*)((char*)ctx + 2048UL * 1024 * 4);

  auto cvt = [&](const float* s, unsigned short* d, long n) {
    long blocks = (n + 255) / 256; if (blocks > 4096) blocks = 4096;
    k_cvt_bf16<<<dim3((unsigned)blocks), dim3(256), 0, stream>>>(s, d, n);
  };
  auto gemm = [&](const unsigned short* A, const unsigned short* Bt, const float* bias,
                  float* C, int M, int N, int K, int batch, long sA, long sB, long sC) {
    k_gemm<<<dim3(N / GBN, M / GBM, batch), dim3(256), 0, stream>>>(A, Bt, bias, C, M, N, K, sA, sB, sC);
  };

  // weights -> bf16
  cvt(Wih_f, Wihf_bf, 2048L * 512);
  cvt(Whh_f, Whhf_bf, 2048L * 512);
  cvt(Wih_b, Wihb_bf, 2048L * 512);
  cvt(Whh_b, Whhb_bf, 2048L * 512);
  cvt(Wih_d, Wihd_bf, 4096L * 512);
  cvt(Whh_d, Whhd_bf, 4096L * 1024);
  cvt(Wq, Wq_bf, 1024L * 1024);
  cvt(Wk, Wk_bf, 1024L * 1024);
  cvt(Wv, Wv_bf, 1024L * 1024);
  cvt(Wfc, Wfc_bf, 32000L * 1024);
  // combined biases bih + bhh
  k_addv<<<dim3(8), dim3(256), 0, stream>>>(bih_f, bhh_f, bias_f, 2048);
  k_addv<<<dim3(8), dim3(256), 0, stream>>>(bih_b, bhh_b, bias_b, 2048);
  k_addv<<<dim3(16), dim3(256), 0, stream>>>(bih_d, bhh_d, bias_d, 4096);
  // embedding gathers
  k_embed<<<dim3(2048), dim3(256), 0, stream>>>(enc_seq, enc_emb, xe_bf, 2048);
  k_embed<<<dim3(2048), dim3(256), 0, stream>>>(dec_seq, dec_emb, xd_bf, 2048);
  // input-gate GEMMs (bias folded in)
  gemm(xe_bf, Wihf_bf, bias_f, xg_f, 2048, 2048, 512, 1, 0, 0, 0);
  gemm(xe_bf, Wihb_bf, bias_b, xg_b, 2048, 2048, 512, 1, 0, 0, 0);
  gemm(xd_bf, Wihd_bf, bias_d, xg_d, 2048, 4096, 512, 1, 0, 0, 0);
  // bidirectional encoder recurrence
  k_enc_lstm<<<dim3(2), dim3(256), 0, stream>>>(xg_f, xg_b, Whhf_bf, Whhb_bf,
                                                gbuf_enc, enc_out, h0buf, c0buf);
  cvt(enc_out, enc_out_bf, 2048L * 1024);
  // decoder recurrence
  k_dec_lstm<<<dim3(1), dim3(1024), 0, stream>>>(xg_d, Whhd_bf, h0buf, c0buf,
                                                 gbuf_dec, dec_out);
  cvt(dec_out, dec_out_bf, 2048L * 1024);
  // attention projections
  gemm(dec_out_bf, Wq_bf, bq, q_f, 2048, 1024, 1024, 1, 0, 0, 0);
  gemm(enc_out_bf, Wk_bf, bk, k_f, 2048, 1024, 1024, 1, 0, 0, 0);
  gemm(enc_out_bf, Wv_bf, bv, v_f, 2048, 1024, 1024, 1, 0, 0, 0);
  cvt(q_f, q_bf, 2048L * 1024);
  cvt(k_f, k_bf, 2048L * 1024);
  k_vt<<<dim3(2048), dim3(256), 0, stream>>>(v_f, vt_bf);
  // scores: sim[b] = q[b] @ k[b]^T  (batched)
  gemm(q_bf, k_bf, nullptr, sim, 128, 128, 1024, 16, 128L * 1024, 128L * 1024, 128L * 128);
  k_softmax<<<dim3(2048), dim3(128), 0, stream>>>(sim, enc_seq, attn_bf);
  // ctx[b] = attn[b] @ v[b]  (Bt = v^T)
  gemm(attn_bf, vt_bf, nullptr, ctx, 128, 1024, 128, 16, 128L * 128, 1024L * 128, 128L * 1024);
  // tanh + residual + layernorm -> bf16
  k_ln<<<dim3(2048), dim3(256), 0, stream>>>(dec_out, ctx, gamma, beta, out_bf);
  // vocab projection (dominant GEMM, 134 GFLOP)
  gemm(out_bf, Wfc_bf, bfc, out, 2048, 32000, 1024, 1, 0, 0, 0);
}
